// VectorQuantize_37898791420258
// MI455X (gfx1250) — matrix operand order repaired and checked by reading, not compiled, not run on hardware
//
#include <hip/hip_runtime.h>
#include <hip/hip_bf16.h>

// Problem constants (from reference)
#define BB 16
#define DD 256
#define TT 2048
#define KK 1024

// LDS tile row padding: 256 bf16 payload, padded to 264 (132 dwords) so lane n
// starts at bank 4n -> conflict-optimal ds_load_b128 across the wave.
#define ROW_EL 264

typedef __attribute__((ext_vector_type(16))) __bf16 v16bf;
typedef __attribute__((ext_vector_type(8)))  float  v8f;

// ---------------------------------------------------------------------------
// CDNA5 async global->LDS copy (ASYNCcnt-tracked) + waits, via inline asm.
// ---------------------------------------------------------------------------
__device__ __forceinline__ void async_g2l_b128(unsigned int lds_byte_addr,
                                               const void* gaddr) {
    asm volatile("global_load_async_to_lds_b128 %0, %1, off"
                 :: "v"(lds_byte_addr),
                    "v"((unsigned long long)(size_t)gaddr)
                 : "memory");
}
__device__ __forceinline__ void s_wait_async_le4() {
    asm volatile("s_wait_asynccnt 4" ::: "memory");
}
__device__ __forceinline__ void s_wait_async_le0() {
    asm volatile("s_wait_asynccnt 0" ::: "memory");
}

// ---------------------------------------------------------------------------
// Prep 1: split codebook fp32 -> bf16 hi/lo planes (row-major [K][D])
// ---------------------------------------------------------------------------
__global__ void vq_split_codebook(const float* __restrict__ cb,
                                  __bf16* __restrict__ cb_hi,
                                  __bf16* __restrict__ cb_lo) {
    int i = blockIdx.x * blockDim.x + threadIdx.x;
    if (i < KK * DD) {
        float c = cb[i];
        __bf16 hi = (__bf16)c;
        cb_hi[i] = hi;
        cb_lo[i] = (__bf16)(c - (float)hi);
    }
}

// ---------------------------------------------------------------------------
// Prep 2: e2[k] = ||codebook[k]||^2   (one block of 256 threads per k)
// ---------------------------------------------------------------------------
__global__ void vq_e2(const float* __restrict__ cb, float* __restrict__ e2) {
    int k    = blockIdx.x;
    int tid  = threadIdx.x;
    int lane = tid & 31;
    int wave = tid >> 5;
    float v = cb[k * DD + tid];
    float s = v * v;
#pragma unroll
    for (int off = 16; off > 0; off >>= 1)
        s += __shfl_xor(s, off, 32);
    __shared__ float sm[8];
    if (lane == 0) sm[wave] = s;
    __syncthreads();
    if (tid == 0) {
        float t = 0.f;
#pragma unroll
        for (int w = 0; w < 8; ++w) t += sm[w];
        e2[k] = t;
    }
}

// ---------------------------------------------------------------------------
// Prep 3: zero the commit-loss accumulator slot in d_out
// ---------------------------------------------------------------------------
__global__ void vq_zero_loss(float* __restrict__ loss) { *loss = 0.f; }

// ---------------------------------------------------------------------------
// Main fused kernel. Block = 256 threads = 8 waves; each wave owns a 16-row
// t-tile (A fragments for all D=256 in registers, bf16 hi/lo split, reused
// across all 64 K-tiles). Codebook B tiles (16 codes x 256 d, hi+lo = 16KB)
// are staged in LDS once per block via async global->LDS B128 copies,
// double-buffered so the copy of tile nt+1 overlaps the 24 WMMAs of tile nt.
// ---------------------------------------------------------------------------
__global__ __launch_bounds__(256, 1)
void vq_main(const float* __restrict__ x,
             const float* __restrict__ cb,
             const __bf16* __restrict__ cb_hi,
             const __bf16* __restrict__ cb_lo,
             const float* __restrict__ e2,
             float* __restrict__ out_q,
             float* __restrict__ out_idx,
             float* __restrict__ out_loss) {
    const int b     = blockIdx.x >> 4;        // T/128 = 16 tiles per batch
    const int ttile = blockIdx.x & 15;
    const int t0    = ttile * 128;
    const int tid   = threadIdx.x;
    const int wave  = tid >> 5;
    const int lane  = tid & 31;
    const int h     = lane >> 4;              // half-wave selector
    const int n     = lane & 15;              // N / row-within-half index
    const int t0w   = t0 + wave * 16;

    const float* __restrict__ xb = x + (size_t)b * DD * TT;

    // Double-buffered codebook tile: [buf][fmt hi/lo][16 rows * ROW_EL]
    __shared__ __align__(16) __bf16 smem[2][2][16 * ROW_EL];

    // ---- Load A fragments (this lane's row t = t0w + n), bf16 hi/lo split.
    // A-layout (16-bit 16x32): element i -> k = i + 16*h (contiguous, matches B).
    v16bf a_hi[8], a_lo[8];
    const int trow = t0w + n;
#pragma unroll
    for (int s = 0; s < 8; ++s) {
#pragma unroll
        for (int i = 0; i < 16; ++i) {
            int d = s * 32 + i + h * 16;
            float v = xb[(size_t)d * TT + trow];   // lanes vary trow -> coalesced
            __bf16 hi = (__bf16)v;
            a_hi[s][i] = hi;
            a_lo[s][i] = (__bf16)(v - (float)hi);
        }
    }

    // ---- Issue async copy of tile 0 (4 x B128 per thread: 2 chunks x 2 fmts)
    {
        const __bf16* gh = cb_hi;              // tile 0 is contiguous 8KB
        const __bf16* gl = cb_lo;
#pragma unroll
        for (int it = 0; it < 2; ++it) {
            int c   = it * 256 + tid;          // 0..511 16B-chunks
            int row = c >> 5, col = c & 31;
            unsigned int lh = (unsigned int)(size_t)&smem[0][0][0]
                              + (unsigned int)(row * ROW_EL * 2 + col * 16);
            unsigned int ll = (unsigned int)(size_t)&smem[0][1][0]
                              + (unsigned int)(row * ROW_EL * 2 + col * 16);
            async_g2l_b128(lh, gh + row * DD + col * 8);
            async_g2l_b128(ll, gl + row * DD + col * 8);
        }
    }

    // ---- Running argmin over K (score = e2[k] - 2*x.e ; x^2 is row-constant)
    float best_d[8];
    int   best_i[8];
#pragma unroll
    for (int r = 0; r < 8; ++r) { best_d[r] = 3.4e38f; best_i[r] = 0; }

    for (int nt = 0; nt < KK / 16; ++nt) {
        const int cur = nt & 1;

        if (nt < KK / 16 - 1) {
            // Prefetch next tile into the other buffer (async, overlaps WMMA)
            const __bf16* gh = cb_hi + (size_t)(nt + 1) * 16 * DD;
            const __bf16* gl = cb_lo + (size_t)(nt + 1) * 16 * DD;
#pragma unroll
            for (int it = 0; it < 2; ++it) {
                int c   = it * 256 + tid;
                int row = c >> 5, col = c & 31;
                unsigned int lh = (unsigned int)(size_t)&smem[cur ^ 1][0][0]
                                  + (unsigned int)(row * ROW_EL * 2 + col * 16);
                unsigned int ll = (unsigned int)(size_t)&smem[cur ^ 1][1][0]
                                  + (unsigned int)(row * ROW_EL * 2 + col * 16);
                async_g2l_b128(lh, gh + row * DD + col * 8);
                async_g2l_b128(ll, gl + row * DD + col * 8);
            }
            s_wait_async_le4();   // async loads complete in order: cur tile done
        } else {
            s_wait_async_le0();
        }
        __syncthreads();          // cur tile visible to all waves

        // B-fragment base for this lane: row n, d-offset 16*h (32B vectors)
        const __bf16* bh_base = &smem[cur][0][n * ROW_EL + h * 16];
        const __bf16* bl_base = &smem[cur][1][n * ROW_EL + h * 16];

        v8f acc0 = {};   // hi*hi
        v8f acc1 = {};   // lo*hi + hi*lo (independent accumulation chain)
#pragma unroll
        for (int s = 0; s < 8; ++s) {
            v16bf bh = *(const v16bf*)(bh_base + s * 32);
            v16bf bl = *(const v16bf*)(bl_base + s * 32);
            acc0 = __builtin_amdgcn_wmma_f32_16x16x32_bf16(
                false, a_hi[s], false, bh, (short)0, acc0, false, false);
            acc1 = __builtin_amdgcn_wmma_f32_16x16x32_bf16(
                false, a_lo[s], false, bh, (short)0, acc1, false, false);
            acc1 = __builtin_amdgcn_wmma_f32_16x16x32_bf16(
                false, a_hi[s], false, bl, (short)0, acc1, false, false);
        }

        const int   kcode = nt * 16 + n;
        const float e2v   = e2[kcode];
#pragma unroll
        for (int r = 0; r < 8; ++r) {
            float score = e2v - 2.0f * (acc0[r] + acc1[r]);
            if (score < best_d[r]) { best_d[r] = score; best_i[r] = kcode; }
        }
        __syncthreads();          // done reading buf[cur] before it is refilled
    }

    // ---- Butterfly reduce over the 16 n-lanes of each half-wave.
    // D-layout: lane L, vgpr r -> (m = r + 8*h, n = L&15). Tie -> lower index.
#pragma unroll
    for (int off = 1; off <= 8; off <<= 1) {
#pragma unroll
        for (int r = 0; r < 8; ++r) {
            float od = __shfl_xor(best_d[r], off, 32);
            int   oi = __shfl_xor(best_i[r], off, 32);
            if (od < best_d[r] || (od == best_d[r] && oi < best_i[r])) {
                best_d[r] = od; best_i[r] = oi;
            }
        }
    }

    __shared__ int idx_lds[128];
    if (n == 0) {                              // lanes 0 and 16 of each wave
#pragma unroll
        for (int r = 0; r < 8; ++r)
            idx_lds[wave * 16 + h * 8 + r] = best_i[r];
    }
    __syncthreads();

    // indices output (as float, per harness output dtype)
    if (tid < 128)
        out_idx[(size_t)b * TT + t0 + tid] = (float)idx_lds[tid];

    // ---- Output phase: quantized = codebook[idx] (STE forward == code),
    // written coalesced along t; fused commit-loss accumulation.
    float lsum = 0.f;
#pragma unroll 4
    for (int it = 0; it < 128; ++it) {
        int e  = it * 256 + tid;               // 128 t x 256 d elements
        int tl = e & 127;
        int d  = e >> 7;
        int ki = idx_lds[tl];
        float q  = cb[(size_t)ki * DD + d];    // L2-resident gather
        int   t  = t0 + tl;
        float xv = xb[(size_t)d * TT + t];
        out_q[(size_t)b * DD * TT + (size_t)d * TT + t] = q;
        float df = xv - q;
        lsum += df * df;
    }
    // block reduce + one atomic per block
#pragma unroll
    for (int off = 16; off > 0; off >>= 1)
        lsum += __shfl_xor(lsum, off, 32);
    __shared__ float red[8];
    if (lane == 0) red[wave] = lsum;
    __syncthreads();
    if (tid == 0) {
        float tot = 0.f;
#pragma unroll
        for (int w = 0; w < 8; ++w) tot += red[w];
        atomicAdd(out_loss, tot * (1.0f / ((float)BB * DD * TT)));
    }
}

// ---------------------------------------------------------------------------
extern "C" void kernel_launch(void* const* d_in, const int* in_sizes, int n_in,
                              void* d_out, int out_size, void* d_ws, size_t ws_size,
                              hipStream_t stream) {
    const float* x  = (const float*)d_in[0];   // [B, D, T] fp32
    const float* cb = (const float*)d_in[1];   // [K, D]    fp32

    // workspace layout
    char* ws = (char*)d_ws;
    __bf16* cb_hi = (__bf16*)ws;                                  // K*D*2 B
    __bf16* cb_lo = (__bf16*)(ws + (size_t)KK * DD * 2);          // K*D*2 B
    float*  e2    = (float*)(ws + (size_t)KK * DD * 4);           // K*4   B

    // output layout: quantized [B,D,T] | indices [B,T] (as float) | loss [1]
    float* out_q    = (float*)d_out;
    float* out_idx  = out_q + (size_t)BB * DD * TT;
    float* out_loss = out_idx + (size_t)BB * TT;

    vq_split_codebook<<<(KK * DD + 255) / 256, 256, 0, stream>>>(cb, cb_hi, cb_lo);
    vq_e2<<<KK, 256, 0, stream>>>(cb, e2);
    vq_zero_loss<<<1, 1, 0, stream>>>(out_loss);
    vq_main<<<BB * (TT / 128), 256, 0, stream>>>(x, cb, cb_hi, cb_lo, e2,
                                                 out_q, out_idx, out_loss);
}
